// GATv2_21139829031699
// MI455X (gfx1250) — compile-verified, matrix operand order checked by reading, not run on hardware
//
#include <hip/hip_runtime.h>
#include <hip/hip_bf16.h>

typedef __attribute__((ext_vector_type(16))) _Float16 v16h;
typedef __attribute__((ext_vector_type(8)))  float    v8f;

#define N_NODES 20000
#define E_EDGES 320000
#define E_TOT   (E_EDGES + N_NODES)   // 340000
#define IN_CH   128
#define HEADS   8
#define CPH     128
#define HC_     1024
#define G_GR    64

// ---------------------------------------------------------------- utilities
__global__ void f32_to_f16_kernel(const float* __restrict__ in,
                                  _Float16* __restrict__ out, int n) {
  int i = blockIdx.x * blockDim.x + threadIdx.x;
  if (i < n) out[i] = (_Float16)in[i];
}

__global__ void init_kernel(float* __restrict__ mmax, float* __restrict__ denom,
                            float* __restrict__ pooled, float* __restrict__ cnt) {
  int i = blockIdx.x * blockDim.x + threadIdx.x;
  if (i < N_NODES * HEADS) { mmax[i] = -3.0e38f; denom[i] = 0.0f; }
  if (i < G_GR * CPH) pooled[i] = 0.0f;
  if (i < G_GR) cnt[i] = 0.0f;
}

__device__ inline void atomicMaxF(float* addr, float v) {
  int* ai = (int*)addr;
  int old = __float_as_int(*addr);
  while (__int_as_float(old) < v) {
    int prev = atomicCAS(ai, old, __float_as_int(v));
    if (prev == old) break;
    old = prev;
  }
}

// ---------------------------------------------------------------- WMMA GEMM
// Out[M,Nout] = A[M,K] * B[Nout,K]^T + bias;  act: 0 = none, 1 = ELU.
// One wave per 16x16 tile, 8 waves / block. K multiple of 32; M,Nout mult of 16.
// A fragment lane layout (ISA 7.12.2, 16-bit A 16x32): lane m(0..15) row m,
// K in {0-7,16-23}; lanes 16-31 same rows, K in {8-15,24-31}; B mirrored on cols.
__global__ __launch_bounds__(256)
void gemm_wmma_kernel(const _Float16* __restrict__ A, const _Float16* __restrict__ B,
                      const float* __restrict__ bias, float* __restrict__ Out,
                      int M, int Nout, int K, int act) {
  int wave = threadIdx.x >> 5;
  int lane = threadIdx.x & 31;
  int tile = blockIdx.x * 8 + wave;
  int tilesN = Nout >> 4;
  int totalTiles = (M >> 4) * tilesN;
  if (tile >= totalTiles) return;          // wave-uniform: EXEC all-ones inside
  int m0 = (tile / tilesN) << 4;
  int n0 = (tile % tilesN) << 4;
  int laneHi = lane >> 4;
  int lmod = lane & 15;

  const _Float16* Arow = A + (size_t)(m0 + lmod) * K;
  const _Float16* Brow = B + (size_t)(n0 + lmod) * K;

  v8f c = {};
  for (int kb = 0; kb < K; kb += 32) {
    __builtin_prefetch(Arow + kb + 64, 0, 1);   // global_prefetch_b8
    __builtin_prefetch(Brow + kb + 64, 0, 1);
    v16h a, b;
#pragma unroll
    for (int v = 0; v < 8; ++v) {
      int kk = kb + ((v >> 2) << 4) + (laneHi << 3) + ((v & 3) << 1);
      a[2 * v]     = Arow[kk];
      a[2 * v + 1] = Arow[kk + 1];
      b[2 * v]     = Brow[kk];
      b[2 * v + 1] = Brow[kk + 1];
    }
    c = __builtin_amdgcn_wmma_f32_16x16x32_f16(false, a, false, b,
                                               (short)0, c, false, false);
  }

  int col = n0 + lmod;
  float bv = bias ? bias[col] : 0.0f;
#pragma unroll
  for (int r = 0; r < 8; ++r) {
    int row = m0 + r + (laneHi << 3);         // D layout: VGPR r -> rows r, r+8
    float v = c[r] + bv;
    if (act == 1) v = (v > 0.0f) ? v : (expf(v) - 1.0f);   // ELU
    Out[(size_t)row * Nout + col] = v;
  }
}

// ---------------------------------------------------------------- edge stage
// Wave per edge: 32-lane contiguous channel chunks (L2-resident gathers).
__global__ __launch_bounds__(256)
void edge_logits_kernel(const int* __restrict__ ei, const float* __restrict__ xl,
                        const float* __restrict__ xr, const float* __restrict__ att,
                        float* __restrict__ logits, float* __restrict__ mmax) {
  int wave = threadIdx.x >> 5, lane = threadIdx.x & 31;
  int e = blockIdx.x * 8 + wave;
  if (e >= E_TOT) return;
  int s, d;
  if (e < E_EDGES) { s = ei[e]; d = ei[E_EDGES + e]; } else { s = d = e - E_EDGES; }
  const float* xls = xl + (size_t)s * HC_;
  const float* xrd = xr + (size_t)d * HC_;
#pragma unroll
  for (int h = 0; h < HEADS; ++h) {
    float acc = 0.0f;
#pragma unroll
    for (int j = 0; j < 4; ++j) {
      int cidx = h * CPH + j * 32 + lane;
      float v = xls[cidx] + xrd[cidx];
      v = (v > 0.0f) ? v : 0.2f * v;          // leaky_relu(0.2)
      acc += v * att[cidx];
    }
    for (int off = 16; off > 0; off >>= 1) acc += __shfl_down(acc, off, 32);
    if (lane == 0) {
      logits[(size_t)e * HEADS + h] = acc;
      atomicMaxF(&mmax[d * HEADS + h], acc);
    }
  }
}

__global__ void edge_exp_kernel(const int* __restrict__ ei, float* __restrict__ logits,
                                const float* __restrict__ mmax, float* __restrict__ denom) {
  int idx = blockIdx.x * blockDim.x + threadIdx.x;
  if (idx >= E_TOT * HEADS) return;
  int e = idx >> 3, h = idx & 7;
  int d = (e < E_EDGES) ? ei[E_EDGES + e] : (e - E_EDGES);
  float a = expf(logits[idx] - mmax[d * HEADS + h]);
  logits[idx] = a;                             // overwrite with numerator
  atomicAdd(&denom[d * HEADS + h], a);
}

__global__ __launch_bounds__(256)
void scatter_kernel(const int* __restrict__ ei, const float* __restrict__ xl,
                    const float* __restrict__ numer, const float* __restrict__ denom,
                    float* __restrict__ out) {
  int wave = threadIdx.x >> 5, lane = threadIdx.x & 31;
  int e = blockIdx.x * 8 + wave;
  if (e >= E_TOT) return;
  int s, d;
  if (e < E_EDGES) { s = ei[e]; d = ei[E_EDGES + e]; } else { s = d = e - E_EDGES; }
  const float* xls = xl + (size_t)s * HC_;
  float* od = out + (size_t)d * HC_;
#pragma unroll
  for (int h = 0; h < HEADS; ++h) {
    float al = numer[(size_t)e * HEADS + h] / (denom[d * HEADS + h] + 1e-16f);
#pragma unroll
    for (int j = 0; j < 4; ++j) {
      int cidx = h * CPH + j * 32 + lane;
      atomicAdd(&od[cidx], xls[cidx] * al);
    }
  }
}

// ---------------------------------------------------------------- pool + MLP
__global__ void pool_kernel(const float* __restrict__ h, const int* __restrict__ batch,
                            float* __restrict__ pooled, float* __restrict__ cnt) {
  int idx = blockIdx.x * blockDim.x + threadIdx.x;
  if (idx >= N_NODES * CPH) return;
  int n = idx / CPH, c = idx % CPH;
  int g = batch[n];
  atomicAdd(&pooled[g * CPH + c], h[idx]);
  if (c == 0) atomicAdd(&cnt[g], 1.0f);
}

__global__ void mlp_kernel(const float* __restrict__ pooled, const float* __restrict__ cnt,
                           const float* __restrict__ W1, const float* __restrict__ b1,
                           const float* __restrict__ W2, const float* __restrict__ b2,
                           const float* __restrict__ W3, const float* __restrict__ b3,
                           float* __restrict__ out) {
  int g = threadIdx.x;
  if (g >= G_GR) return;
  float inv = 1.0f / fmaxf(cnt[g], 1.0f);
  float y1[16];
  for (int i = 0; i < 16; ++i) {
    float s = b1[i];
    for (int k = 0; k < CPH; ++k) s += W1[i * CPH + k] * pooled[g * CPH + k] * inv;
    y1[i] = fmaxf(s, 0.0f);
  }
  float y2[32];
  for (int i = 0; i < 32; ++i) {
    float s = b2[i];
    for (int k = 0; k < 16; ++k) s += W2[i * 16 + k] * y1[k];
    y2[i] = fmaxf(s, 0.0f);
  }
  for (int i = 0; i < 5; ++i) {
    float s = b3[i];
    for (int k = 0; k < 32; ++k) s += W3[i * 32 + k] * y2[k];
    out[g * 5 + i] = s;
  }
}

// ---------------------------------------------------------------- launcher
static inline size_t alignup(size_t x) { return (x + 255) & ~(size_t)255; }

extern "C" void kernel_launch(void* const* d_in, const int* in_sizes, int n_in,
                              void* d_out, int out_size, void* d_ws, size_t ws_size,
                              hipStream_t stream) {
  const float* x        = (const float*)d_in[0];
  const int*   ei       = (const int*)d_in[1];   // [2,E] flat
  const int*   batch    = (const int*)d_in[2];
  const float* Wl       = (const float*)d_in[3];
  const float* bl       = (const float*)d_in[4];
  const float* Wr       = (const float*)d_in[5];
  const float* br       = (const float*)d_in[6];
  const float* att      = (const float*)d_in[7];
  const float* Wres     = (const float*)d_in[8];
  const float* bias_conv= (const float*)d_in[9];
  const float* Wlin     = (const float*)d_in[10];
  const float* blin     = (const float*)d_in[11];
  const float* W1       = (const float*)d_in[12];
  const float* b1       = (const float*)d_in[13];
  const float* W2       = (const float*)d_in[14];
  const float* b2       = (const float*)d_in[15];
  const float* W3       = (const float*)d_in[16];
  const float* b3       = (const float*)d_in[17];
  float* out = (float*)d_out;

  // workspace carve-out
  char* p = (char*)d_ws;
  size_t off = 0;
  _Float16* xh    = (_Float16*)(p + off); off = alignup(off + (size_t)N_NODES*IN_CH*2);
  _Float16* Wlh   = (_Float16*)(p + off); off = alignup(off + (size_t)HC_*IN_CH*2);
  _Float16* Wrh   = (_Float16*)(p + off); off = alignup(off + (size_t)HC_*IN_CH*2);
  _Float16* Wresh = (_Float16*)(p + off); off = alignup(off + (size_t)HC_*IN_CH*2);
  _Float16* Wlinh = (_Float16*)(p + off); off = alignup(off + (size_t)CPH*HC_*2);
  float* x_l      = (float*)(p + off);    off = alignup(off + (size_t)N_NODES*HC_*4);
  float* x_r      = (float*)(p + off);    off = alignup(off + (size_t)N_NODES*HC_*4);
  float* out_agg  = (float*)(p + off);    off = alignup(off + (size_t)N_NODES*HC_*4);
  float* logits   = (float*)(p + off);    off = alignup(off + (size_t)E_TOT*HEADS*4);
  float* mmax     = (float*)(p + off);    off = alignup(off + (size_t)N_NODES*HEADS*4);
  float* denom    = (float*)(p + off);    off = alignup(off + (size_t)N_NODES*HEADS*4);
  float* hbuf     = (float*)(p + off);    off = alignup(off + (size_t)N_NODES*CPH*4);
  float* pooled   = (float*)(p + off);    off = alignup(off + (size_t)G_GR*CPH*4);
  float* cnt      = (float*)(p + off);    off = alignup(off + (size_t)G_GR*4);
  // totalh (f16 input to post-GEMM) aliases x_r: x_r dead after edge_logits,
  // totalh written only after scatter completes.
  _Float16* totalh = (_Float16*)x_r;

  const int T = 256;
  // 1. f16 conversions
  f32_to_f16_kernel<<<(N_NODES*IN_CH + T-1)/T, T, 0, stream>>>(x,    xh,    N_NODES*IN_CH);
  f32_to_f16_kernel<<<(HC_*IN_CH    + T-1)/T, T, 0, stream>>>(Wl,   Wlh,   HC_*IN_CH);
  f32_to_f16_kernel<<<(HC_*IN_CH    + T-1)/T, T, 0, stream>>>(Wr,   Wrh,   HC_*IN_CH);
  f32_to_f16_kernel<<<(HC_*IN_CH    + T-1)/T, T, 0, stream>>>(Wres, Wresh, HC_*IN_CH);
  f32_to_f16_kernel<<<(CPH*HC_      + T-1)/T, T, 0, stream>>>(Wlin, Wlinh, CPH*HC_);
  // 2. init softmax/pool state
  init_kernel<<<(N_NODES*HEADS + T-1)/T, T, 0, stream>>>(mmax, denom, pooled, cnt);
  // 3. input GEMMs on WMMA: tiles = (20000/16)*(1024/16) = 80000 -> 10000 blocks
  gemm_wmma_kernel<<<10000, T, 0, stream>>>(xh, Wlh,   bl,        x_l,     N_NODES, HC_, IN_CH, 0);
  gemm_wmma_kernel<<<10000, T, 0, stream>>>(xh, Wrh,   br,        x_r,     N_NODES, HC_, IN_CH, 0);
  gemm_wmma_kernel<<<10000, T, 0, stream>>>(xh, Wresh, bias_conv, out_agg, N_NODES, HC_, IN_CH, 0);
  // 4-6. edge stage (wave per edge; E_TOT = 340000 = 42500 * 8 waves)
  edge_logits_kernel<<<E_TOT/8, T, 0, stream>>>(ei, x_l, x_r, att, logits, mmax);
  edge_exp_kernel<<<(E_TOT*HEADS + T-1)/T, T, 0, stream>>>(ei, logits, mmax, denom);
  scatter_kernel<<<E_TOT/8, T, 0, stream>>>(ei, x_l, logits, denom, out_agg);
  // 7. convert accumulated (residual+bias+messages) to f16
  f32_to_f16_kernel<<<(N_NODES*HC_ + T-1)/T, T, 0, stream>>>(out_agg, totalh, N_NODES*HC_);
  // 8. Linear(1024->128) + ELU: tiles = 1250*8 = 10000 -> 1250 blocks
  gemm_wmma_kernel<<<1250, T, 0, stream>>>(totalh, Wlinh, blin, hbuf, N_NODES, CPH, HC_, 1);
  // 9. mean pool
  pool_kernel<<<(N_NODES*CPH + T-1)/T, T, 0, stream>>>(hbuf, batch, pooled, cnt);
  // 10. MLP head
  mlp_kernel<<<1, 64, 0, stream>>>(pooled, cnt, W1, b1, W2, b2, W3, b3, out);
}
